// MuliHeadAttention_29540785062464
// MI455X (gfx1250) — matrix operand order repaired and checked by reading, not compiled, not run on hardware
//
#include <hip/hip_runtime.h>

typedef __attribute__((ext_vector_type(16))) _Float16 v16h;
typedef __attribute__((ext_vector_type(8)))  _Float16 v8h;
typedef __attribute__((ext_vector_type(8)))  float    v8f;

constexpr int B_ = 4, S_ = 2048, HID_ = 768, H_ = 6, DK_ = 128;
constexpr float SCALE_ = 0.0883883476483184f; // 128^-0.5

// ---- WMMA fragment helpers (CDNA5 16x16x32 f16 layouts) -------------------
// A (16x32, MxK): lane<16 holds row (lane&15), halfs K{koff..koff+7, koff+16..koff+23}
// with koff = (lane>>4)*8.  Two 16B loads.
__device__ __forceinline__ v16h fragA(const _Float16* base, int stride, int lane) {
  const int r = lane & 15, hi = (lane >> 4) & 1;
  const _Float16* p = base + r * stride + hi * 8;
  v8h lo = *(const v8h*)(p);
  v8h hh = *(const v8h*)(p + 16);
  v16h a;
#pragma unroll
  for (int i = 0; i < 8; ++i) { a[i] = lo[i]; a[i + 8] = hh[i]; }
  return a;
}

// B (32x16, KxN), read from a TRANSPOSED tile Bt[n][k]: lane holds column
// n=(lane&15), K{koff..koff+7, koff+16..koff+23} with koff = hi*8.
__device__ __forceinline__ v16h fragB(const _Float16* base, int stride, int lane) {
  const int n = lane & 15, hi = (lane >> 4) & 1;
  const _Float16* p = base + n * stride + hi * 8;
  v8h lo = *(const v8h*)(p);
  v8h hh = *(const v8h*)(p + 16);
  v16h b;
#pragma unroll
  for (int i = 0; i < 8; ++i) { b[i] = lo[i]; b[i + 8] = hh[i]; }
  return b;
}

__device__ __forceinline__ v8f wmma_f16(v16h a, v16h b, v8f c) {
  return __builtin_amdgcn_wmma_f32_16x16x32_f16(false, a, false, b, (short)0, c,
                                                false, false);
}

// ===========================================================================
// Kernel 1: fused Q/K/V projections.  M=8192, N=768, K=768.
// Block = 128 thr (4 waves), tile 64x64, each wave 32x32.  blockIdx.z = proj.
// Output f16, head-major [B,H,S,DK]; Q pre-scaled by 1/sqrt(dk).
// ===========================================================================
constexpr int GSTR = 40; // LDS half-stride (80B, 16B aligned)

__global__ __launch_bounds__(128) void proj_kernel(
    const float* __restrict__ q, const float* __restrict__ k,
    const float* __restrict__ v, const float* __restrict__ Wq,
    const float* __restrict__ Wk, const float* __restrict__ Wv,
    _Float16* __restrict__ Qh, _Float16* __restrict__ Kh,
    _Float16* __restrict__ Vh) {
  __shared__ _Float16 As[64 * GSTR];
  __shared__ _Float16 Bs[64 * GSTR]; // transposed: Bs[n][k]
  const int z = blockIdx.z;
  const float* A = (z == 0) ? q : (z == 1) ? k : v;
  const float* W = (z == 0) ? Wq : (z == 1) ? Wk : Wv;
  _Float16* O = (z == 0) ? Qh : (z == 1) ? Kh : Vh;
  const float scale = (z == 0) ? SCALE_ : 1.0f;
  const int mbase = blockIdx.y * 64, nbase = blockIdx.x * 64;
  const int tid = threadIdx.x, lane = tid & 31, wv = tid >> 5;
  const int wm = (wv >> 1) * 32, wn = (wv & 1) * 32;
  v8f c00 = {0,0,0,0,0,0,0,0}, c01 = c00, c10 = c00, c11 = c00;

  for (int kb = 0; kb < HID_; kb += 32) {
#pragma unroll 4
    for (int i = tid; i < 64 * 32; i += 128) {
      const int r0 = i >> 5, k0 = i & 31;
      As[r0 * GSTR + k0] = (_Float16)A[(size_t)(mbase + r0) * HID_ + kb + k0];
      Bs[r0 * GSTR + k0] = (_Float16)W[(size_t)(kb + k0) * HID_ + nbase + r0];
    }
    __syncthreads();
    v16h a0 = fragA(As + (wm + 0) * GSTR, GSTR, lane);
    v16h a1 = fragA(As + (wm + 16) * GSTR, GSTR, lane);
    v16h b0 = fragB(Bs + (wn + 0) * GSTR, GSTR, lane);
    v16h b1 = fragB(Bs + (wn + 16) * GSTR, GSTR, lane);
    c00 = wmma_f16(a0, b0, c00);
    c01 = wmma_f16(a0, b1, c01);
    c10 = wmma_f16(a1, b0, c10);
    c11 = wmma_f16(a1, b1, c11);
    __syncthreads();
  }
  const int r = lane & 15, hi = lane >> 4;
#pragma unroll
  for (int ci = 0; ci < 2; ++ci)
#pragma unroll
    for (int cj = 0; cj < 2; ++cj) {
      v8f cc = (ci == 0) ? ((cj == 0) ? c00 : c01) : ((cj == 0) ? c10 : c11);
#pragma unroll
      for (int e = 0; e < 8; ++e) {
        const int m = mbase + wm + ci * 16 + hi * 8 + e;
        const int n = nbase + wn + cj * 16 + r;
        const int bb = m >> 11, ss = m & (S_ - 1);
        const int hd = n >> 7, dk = n & (DK_ - 1);
        O[(((size_t)bb * H_ + hd) * S_ + ss) * DK_ + dk] =
            (_Float16)(cc[e] * scale);
      }
    }
}

// ===========================================================================
// Kernel 2: flash attention with adjacency gate + mask.
// Grid (S/64, H, B), block 128 (4 waves).  Wave w owns 16 q-rows.
// KV tiles of 64; scores/PV via f16 WMMA, online softmax in f32.
// ===========================================================================
constexpr int KSTR = 136; // 272B rows, 16B aligned
constexpr int VSTR = 72;  // 144B rows
constexpr int PSTR = 72;

__global__ __launch_bounds__(128) void attn_kernel(
    const _Float16* __restrict__ Qh, const _Float16* __restrict__ Kh,
    const _Float16* __restrict__ Vh, const float* __restrict__ adj,
    const unsigned char* __restrict__ msk, _Float16* __restrict__ AttO) {
  __shared__ _Float16 Kt[64 * KSTR];      // K tile, row-major [kv][d]
  __shared__ _Float16 Vt[128 * VSTR];     // V tile, transposed [d][kv]
  __shared__ _Float16 Pl[4 * 16 * PSTR];  // per-wave P staging

  const int qt = blockIdx.x, h = blockIdx.y, b = blockIdx.z;
  const int qbase = qt * 64;
  const size_t hoff = (size_t)(b * H_ + h) * S_ * DK_;
  const _Float16* Qp = Qh + hoff;
  const _Float16* Kp = Kh + hoff;
  const _Float16* Vp = Vh + hoff;
  const float* adjb = adj + (size_t)b * S_ * S_;
  const unsigned char* mb = msk + (size_t)b * S_ * S_;
  const int tid = threadIdx.x, lane = tid & 31, w = tid >> 5;
  const int r = lane & 15, hi = lane >> 4;

  // Q fragments for this wave's 16 rows (already scaled), K-dim = 128 = 4 steps
  v16h qf[4];
  {
    const _Float16* qrow = Qp + (size_t)(qbase + w * 16) * DK_;
#pragma unroll
    for (int kk = 0; kk < 4; ++kk) qf[kk] = fragA(qrow + kk * 32, DK_, lane);
  }

  v8f acc[8];
  float mrun[8], lrun[8];
#pragma unroll
  for (int n = 0; n < 8; ++n) acc[n] = v8f{0,0,0,0,0,0,0,0};
#pragma unroll
  for (int e = 0; e < 8; ++e) { mrun[e] = -3.0e38f; lrun[e] = 0.0f; }

  for (int kv = 0; kv < S_; kv += 64) {
    // cooperative K tile load (vectorized 16B) and V transpose load
#pragma unroll 2
    for (int i = tid; i < 64 * 16; i += 128) {
      const int rr = i >> 4, cc = (i & 15) * 8;
      *(v8h*)&Kt[rr * KSTR + cc] = *(const v8h*)&Kp[(size_t)(kv + rr) * DK_ + cc];
    }
    for (int i = tid; i < 64 * 128; i += 128) {
      const int rr = i >> 7, dd = i & 127;
      Vt[dd * VSTR + rr] = Vp[(size_t)(kv + rr) * DK_ + dd];
    }
    if (kv + 64 < S_) {
      __builtin_prefetch(&Kp[(size_t)(kv + 64) * DK_], 0, 1);
      __builtin_prefetch(&Vp[(size_t)(kv + 64) * DK_], 0, 1);
    }
    __syncthreads();

    // scores: 16 x 64 tile = 4 n-tiles x 4 k-steps of WMMA
    v8f sf[4];
#pragma unroll
    for (int j = 0; j < 4; ++j) {
      v8f s = v8f{0,0,0,0,0,0,0,0};
#pragma unroll
      for (int kk = 0; kk < 4; ++kk) {
        v16h bf = fragB(Kt + (j * 16) * KSTR + kk * 32, KSTR, lane);
        s = wmma_f16(qf[kk], bf, s);
      }
      sf[j] = s;
    }

    // adjacency gate + mask (C layout: row = hi*8+e, col = j*16 + r)
#pragma unroll
    for (int j = 0; j < 4; ++j) {
      const int kcol = kv + j * 16 + r;
#pragma unroll
      for (int e = 0; e < 8; ++e) {
        const int qrow = qbase + w * 16 + hi * 8 + e;
        const size_t idx = (size_t)qrow * S_ + kcol;
        const float gated = sf[j][e] * adjb[idx];
        sf[j][e] = mb[idx] ? -1.0e9f : gated;
      }
    }

    // online softmax: reduce across lanes 0-15 / 16-31 (row lives across 16 lanes)
#pragma unroll
    for (int e = 0; e < 8; ++e) {
      float mx = fmaxf(fmaxf(sf[0][e], sf[1][e]), fmaxf(sf[2][e], sf[3][e]));
#pragma unroll
      for (int d = 1; d < 16; d <<= 1) mx = fmaxf(mx, __shfl_xor(mx, d, 16));
      const float mnew = fmaxf(mrun[e], mx);
      const float scl = __expf(mrun[e] - mnew);
      float rs = 0.0f;
#pragma unroll
      for (int j = 0; j < 4; ++j) {
        const float p = __expf(sf[j][e] - mnew);
        sf[j][e] = p;
        rs += p;
      }
#pragma unroll
      for (int d = 1; d < 16; d <<= 1) rs += __shfl_xor(rs, d, 16);
      lrun[e] = lrun[e] * scl + rs;
      mrun[e] = mnew;
#pragma unroll
      for (int n = 0; n < 8; ++n) acc[n][e] *= scl;
    }

    // stage P (C layout -> row-major LDS -> A layout)
    _Float16* Pw = Pl + w * 16 * PSTR;
#pragma unroll
    for (int j = 0; j < 4; ++j)
#pragma unroll
      for (int e = 0; e < 8; ++e)
        Pw[(hi * 8 + e) * PSTR + j * 16 + r] = (_Float16)sf[j][e];
    asm volatile("s_wait_dscnt 0" ::: "memory");

    // O += P(16x64) * V(64x128): 8 n-tiles x 2 k-steps
#pragma unroll
    for (int n = 0; n < 8; ++n) {
#pragma unroll
      for (int kk = 0; kk < 2; ++kk) {
        v16h pf = fragA(Pw + kk * 32, PSTR, lane);
        v16h vf = fragB(Vt + (n * 16) * VSTR + kk * 32, VSTR, lane);
        acc[n] = wmma_f16(pf, vf, acc[n]);
      }
    }
    __syncthreads();
  }

  // epilogue: normalize and store head-concat f16 [B,S,HID]
#pragma unroll
  for (int n = 0; n < 8; ++n) {
#pragma unroll
    for (int e = 0; e < 8; ++e) {
      const int qrow = qbase + w * 16 + hi * 8 + e;
      const float val = acc[n][e] / lrun[e];
      AttO[((size_t)(b * S_ + qrow)) * HID_ + h * DK_ + n * 16 + r] =
          (_Float16)val;
    }
  }
}

// ===========================================================================
// Kernel 3: output projection, f16 [8192x768] x f32 Wo -> f32 out.
// ===========================================================================
__global__ __launch_bounds__(128) void oproj_kernel(
    const _Float16* __restrict__ A, const float* __restrict__ W,
    float* __restrict__ out) {
  __shared__ _Float16 As[64 * GSTR];
  __shared__ _Float16 Bs[64 * GSTR];
  const int mbase = blockIdx.y * 64, nbase = blockIdx.x * 64;
  const int tid = threadIdx.x, lane = tid & 31, wv = tid >> 5;
  const int wm = (wv >> 1) * 32, wn = (wv & 1) * 32;
  v8f c00 = {0,0,0,0,0,0,0,0}, c01 = c00, c10 = c00, c11 = c00;

  for (int kb = 0; kb < HID_; kb += 32) {
#pragma unroll 4
    for (int i = tid; i < 64 * 32; i += 128) {
      const int r0 = i >> 5, k0 = i & 31;
      As[r0 * GSTR + k0] = A[(size_t)(mbase + r0) * HID_ + kb + k0];
      Bs[r0 * GSTR + k0] = (_Float16)W[(size_t)(kb + k0) * HID_ + nbase + r0];
    }
    __syncthreads();
    v16h a0 = fragA(As + (wm + 0) * GSTR, GSTR, lane);
    v16h a1 = fragA(As + (wm + 16) * GSTR, GSTR, lane);
    v16h b0 = fragB(Bs + (wn + 0) * GSTR, GSTR, lane);
    v16h b1 = fragB(Bs + (wn + 16) * GSTR, GSTR, lane);
    c00 = wmma_f16(a0, b0, c00);
    c01 = wmma_f16(a0, b1, c01);
    c10 = wmma_f16(a1, b0, c10);
    c11 = wmma_f16(a1, b1, c11);
    __syncthreads();
  }
  const int r = lane & 15, hi = lane >> 4;
#pragma unroll
  for (int ci = 0; ci < 2; ++ci)
#pragma unroll
    for (int cj = 0; cj < 2; ++cj) {
      v8f cc = (ci == 0) ? ((cj == 0) ? c00 : c01) : ((cj == 0) ? c10 : c11);
#pragma unroll
      for (int e = 0; e < 8; ++e) {
        const int m = mbase + wm + ci * 16 + hi * 8 + e;
        const int n = nbase + wn + cj * 16 + r;
        out[(size_t)m * HID_ + n] = cc[e];
      }
    }
}

// ===========================================================================
extern "C" void kernel_launch(void* const* d_in, const int* in_sizes, int n_in,
                              void* d_out, int out_size, void* d_ws,
                              size_t ws_size, hipStream_t stream) {
  (void)in_sizes; (void)n_in; (void)out_size; (void)ws_size;
  const float* q = (const float*)d_in[0];
  const float* k = (const float*)d_in[1];
  const float* v = (const float*)d_in[2];
  const float* adj = (const float*)d_in[3];
  const unsigned char* mask = (const unsigned char*)d_in[4];
  const float* Wq = (const float*)d_in[5];
  const float* Wk = (const float*)d_in[6];
  const float* Wv = (const float*)d_in[7];
  const float* Wo = (const float*)d_in[8];
  float* out = (float*)d_out;

  const size_t per = (size_t)B_ * H_ * S_ * DK_; // elements per projection
  _Float16* Qh = (_Float16*)d_ws;
  _Float16* Kh = Qh + per;
  _Float16* Vh = Kh + per;
  _Float16* Ao = Vh + per; // [B,S,HID] f16

  dim3 g1(HID_ / 64, (B_ * S_) / 64, 3);
  proj_kernel<<<g1, 128, 0, stream>>>(q, k, v, Wq, Wk, Wv, Qh, Kh, Vh);

  dim3 g2(S_ / 64, H_, B_);
  attn_kernel<<<g2, 128, 0, stream>>>(Qh, Kh, Vh, adj, mask, Ao);

  dim3 g3(HID_ / 64, (B_ * S_) / 64, 1);
  oproj_kernel<<<g3, 128, 0, stream>>>(Ao, Wo, out);
}
